// SelfAttention_71287867179099
// MI455X (gfx1250) — matrix operand order repaired and checked by reading, not compiled, not run on hardware
//
#include <hip/hip_runtime.h>
#include <cstdint>

#define SEQ   4096
#define DIM   1024
#define BM    128
#define BN    128
#define BK    32
#define LDT   40        // padded LDS row stride in bf16 elems (32 + 8) -> conflict-free b128 reads
#define USE_ASYNC 1     // CDNA5 GLOBAL_LOAD_ASYNC_TO_LDS for pure-bf16 staging

typedef unsigned short u16;
typedef __attribute__((ext_vector_type(16))) __bf16 v16bf;
typedef __attribute__((ext_vector_type(8)))  float  v8f;

union Frag { v16bf v; uint4 u[2]; };

// ---------- scalar helpers ----------
__device__ __forceinline__ u16 f32_to_bf16(float f) {
  unsigned u = __float_as_uint(f);
  u += 0x7fffu + ((u >> 16) & 1u);            // round-to-nearest-even
  return (u16)(u >> 16);
}
__device__ __forceinline__ float bf16_to_f32(u16 h) {
  return __uint_as_float(((unsigned)h) << 16);
}
__device__ __forceinline__ unsigned pack2(float a, float b) {
  return (unsigned)f32_to_bf16(a) | ((unsigned)f32_to_bf16(b) << 16);
}
// order-preserving float<->uint encoding so atomicMax(uint) == float max (deterministic)
__device__ __forceinline__ unsigned f2ord(float f) {
  unsigned u = __float_as_uint(f);
  return (u & 0x80000000u) ? ~u : (u | 0x80000000u);
}
__device__ __forceinline__ float ord2f(unsigned u) {
  unsigned b = (u & 0x80000000u) ? (u ^ 0x80000000u) : ~u;
  return __uint_as_float(b);
}

// ---------- CDNA5 async memory->LDS copy (ASYNCcnt tracked) ----------
__device__ __forceinline__ void async_b128(const u16* g, u16* l) {
#if USE_ASYNC
  unsigned lds = (unsigned)(unsigned long long)l;   // low 32 bits of generic ptr = LDS offset
  unsigned long long ga = (unsigned long long)g;
  asm volatile("global_load_async_to_lds_b128 %0, %1, off"
               :: "v"(lds), "v"(ga) : "memory");
#else
  *(uint4*)l = *(const uint4*)g;
#endif
}
__device__ __forceinline__ void wait_async0() {
#if USE_ASYNC
  asm volatile("s_wait_asynccnt 0" ::: "memory");
#endif
}

// ---------- staging: 128x32 bf16 tile, 256 threads, 16 elems (32B) per thread ----------
__device__ __forceinline__ void stage_async(const u16* __restrict__ g, long long row0, int k0,
                                            int ld, u16* lds, int tid) {
  int r = tid >> 1, h = tid & 1;
  const u16* src = g + (row0 + r) * (long long)ld + k0 + h * 16;
  u16* dst = &lds[r * LDT + h * 16];
  async_b128(src, dst);
  async_b128(src + 8, dst + 8);
}

// fp32 A-path (needs conversion): split global-load / LDS-commit for software pipelining
struct RegsF32 { float4 a, b, c, d; };
__device__ __forceinline__ void ldA_f32(const float* __restrict__ g, long long row0, int k0,
                                        int ld, int tid, RegsF32& R) {
  int r = tid >> 1, h = tid & 1;
  const float4* src = (const float4*)(g + (row0 + r) * (long long)ld + k0 + h * 16);
  R.a = src[0]; R.b = src[1]; R.c = src[2]; R.d = src[3];
}
__device__ __forceinline__ void stA_f32(const RegsF32& R, u16* lds, int tid) {
  int r = tid >> 1, h = tid & 1;
  uint4 lo, hi;
  lo.x = pack2(R.a.x, R.a.y); lo.y = pack2(R.a.z, R.a.w);
  lo.z = pack2(R.b.x, R.b.y); lo.w = pack2(R.b.z, R.b.w);
  hi.x = pack2(R.c.x, R.c.y); hi.y = pack2(R.c.z, R.c.w);
  hi.z = pack2(R.d.x, R.d.y); hi.w = pack2(R.d.z, R.d.w);
  uint4* dst = (uint4*)&lds[r * LDT + h * 16];
  dst[0] = lo; dst[1] = hi;
}

// ---------- fragment loads per documented 16-bit WMMA layouts ----------
// A 16x32: lanes 0-15 hold K=0..7 & 16..23, lanes 16-31 hold K=8..15 & 24..31
__device__ __forceinline__ void load_fa(const u16* As, int wrow, int lane, Frag a[2]) {
  int row  = lane & 15;
  int koff = (lane < 16) ? 0 : 8;
#pragma unroll
  for (int f = 0; f < 2; ++f) {
    const u16* p = &As[(wrow * 32 + f * 16 + row) * LDT + koff];
    a[f].u[0] = *(const uint4*)p;
    a[f].u[1] = *(const uint4*)(p + 16);
  }
}
// B 32x16 stored [N][K] in LDS: lanes 0-15 hold K=0..7 & 16..23, lanes 16-31 hold K=8..15 & 24..31
__device__ __forceinline__ void load_fb(const u16* Bs, int wcol, int lane, Frag b[4]) {
  int col = lane & 15;
  int kb  = (lane < 16) ? 0 : 8;
#pragma unroll
  for (int g = 0; g < 4; ++g) {
    const u16* p = &Bs[(wcol * 64 + g * 16 + col) * LDT + kb];
    b[g].u[0] = *(const uint4*)p;
    b[g].u[1] = *(const uint4*)(p + 16);
  }
}

// ---------- double-buffered WMMA GEMM core: block tile 128x128, 8 waves -> 2x4 frags ----------
template<bool A_IS_F32>
__device__ __forceinline__ void gemm_tiles(const void* Ag, const u16* Bg,
                                           int lda, int ldb, int kdim,
                                           long long m0, long long n0,
                                           u16* As0, u16* Bs0, v8f acc[2][4]) {
  const int tid  = threadIdx.x;
  const int lane = tid & 31;
  const int wid  = tid >> 5;
  const int wrow = wid & 3;
  const int wcol = wid >> 2;
  const int nk   = kdim / BK;

  // prologue: stage tile 0 into buffer 0
  if (A_IS_F32) {
    RegsF32 R;
    ldA_f32((const float*)Ag, m0, 0, lda, tid, R);
    stA_f32(R, As0, tid);
  } else {
    stage_async((const u16*)Ag, m0, 0, lda, As0, tid);
  }
  stage_async(Bg, n0, 0, ldb, Bs0, tid);
  wait_async0();
  __syncthreads();

  for (int k = 0; k < nk; ++k) {
    const int cur = k & 1, nxt = cur ^ 1;
    const u16* As = As0 + cur * (BM * LDT);
    const u16* Bs = Bs0 + cur * (BN * LDT);
    const bool pf = (k + 1 < nk);
    RegsF32 R;
    if (pf) {   // issue staging for tile k+1; stays in flight during WMMA on tile k
      if (A_IS_F32) ldA_f32((const float*)Ag, m0, (k + 1) * BK, lda, tid, R);
      else stage_async((const u16*)Ag, m0, (k + 1) * BK, lda, As0 + nxt * (BM * LDT), tid);
      stage_async(Bg, n0, (k + 1) * BK, ldb, Bs0 + nxt * (BN * LDT), tid);
    }
    Frag fa[2], fb[4];
    load_fa(As, wrow, lane, fa);
    load_fb(Bs, wcol, lane, fb);
#pragma unroll
    for (int f = 0; f < 2; ++f)
#pragma unroll
      for (int g = 0; g < 4; ++g)
        acc[f][g] = __builtin_amdgcn_wmma_f32_16x16x32_bf16(
            false, fa[f].v, false, fb[g].v, (short)0, acc[f][g], false, false);
    if (pf) {
      if (A_IS_F32) stA_f32(R, As0 + nxt * (BM * LDT), tid);
      wait_async0();
    }
    __syncthreads();
  }
}

// ---------- kernels ----------
__global__ void k_convert_w(const float* __restrict__ Wq, const float* __restrict__ Wk,
                            const float* __restrict__ Wv, u16* __restrict__ WtBase) {
  const float* W = (blockIdx.y == 0) ? Wq : (blockIdx.y == 1) ? Wk : Wv;
  u16* Wt = WtBase + (size_t)blockIdx.y * DIM * DIM;
  int t = blockIdx.x * blockDim.x + threadIdx.x;      // coalesced read of W[k][n]
  int k = t >> 10, n = t & 1023;
  Wt[(size_t)n * DIM + k] = f32_to_bf16(W[t]);        // write W^T (bf16, [N][K])
}

__global__ void k_init(unsigned* __restrict__ rowmax) {
  int t = blockIdx.x * blockDim.x + threadIdx.x;
  if (t < SEQ) rowmax[t] = 0u;                        // ordered-encoding: 0 < enc(any float)
}

__global__ void k_qkv(const float* __restrict__ x, const u16* __restrict__ WtBase,
                      u16* __restrict__ Qb, u16* __restrict__ Kb, u16* __restrict__ Vt) {
  __shared__ u16 As[2 * BM * LDT];
  __shared__ u16 Bs[2 * BN * LDT];
  long long m0 = (long long)blockIdx.x * BM;
  long long n0 = (long long)blockIdx.y * BN;
  int z = blockIdx.z;
  const u16* Wt = WtBase + (size_t)z * DIM * DIM;
  v8f acc[2][4];
#pragma unroll
  for (int f = 0; f < 2; ++f)
#pragma unroll
    for (int g = 0; g < 4; ++g) acc[f][g] = (v8f)0.0f;

  gemm_tiles<true>(x, Wt, DIM, DIM, DIM, m0, n0, As, Bs, acc);

  int lane = threadIdx.x & 31, wid = threadIdx.x >> 5;
  int wrow = wid & 3, wcol = wid >> 2;
  if (z < 2) {                                        // Q / K row-major bf16
    u16* out = z ? Kb : Qb;
#pragma unroll
    for (int f = 0; f < 2; ++f) {
      long long mb = m0 + wrow * 32 + f * 16 + ((lane < 16) ? 0 : 8);
#pragma unroll
      for (int g = 0; g < 4; ++g) {
        long long n = n0 + wcol * 64 + g * 16 + (lane & 15);
#pragma unroll
        for (int i = 0; i < 8; ++i)
          out[(mb + i) * DIM + n] = f32_to_bf16(acc[f][g][i]);
      }
    }
  } else {                                            // V^T bf16: lane's 8 M-values contiguous
#pragma unroll
    for (int f = 0; f < 2; ++f) {
      long long mb = m0 + wrow * 32 + f * 16 + ((lane < 16) ? 0 : 8);
#pragma unroll
      for (int g = 0; g < 4; ++g) {
        long long n = n0 + wcol * 64 + g * 16 + (lane & 15);
        uint4 p;
        p.x = pack2(acc[f][g][0], acc[f][g][1]);
        p.y = pack2(acc[f][g][2], acc[f][g][3]);
        p.z = pack2(acc[f][g][4], acc[f][g][5]);
        p.w = pack2(acc[f][g][6], acc[f][g][7]);
        *(uint4*)&Vt[n * (long long)SEQ + mb] = p;
      }
    }
  }
}

__global__ void k_scores(const u16* __restrict__ Qb, const u16* __restrict__ Kb,
                         u16* __restrict__ Sc, unsigned* __restrict__ rowmax) {
  __shared__ u16 As[2 * BM * LDT];
  __shared__ u16 Bs[2 * BN * LDT];
  long long m0 = (long long)blockIdx.x * BM;
  long long n0 = (long long)blockIdx.y * BN;
  v8f acc[2][4];
#pragma unroll
  for (int f = 0; f < 2; ++f)
#pragma unroll
    for (int g = 0; g < 4; ++g) acc[f][g] = (v8f)0.0f;

  gemm_tiles<false>(Qb, Kb, DIM, DIM, DIM, m0, n0, As, Bs, acc);

  const float scale = 0.03125f;                       // 1/sqrt(1024)
  int lane = threadIdx.x & 31, wid = threadIdx.x >> 5;
  int wrow = wid & 3, wcol = wid >> 2;
#pragma unroll
  for (int f = 0; f < 2; ++f) {
#pragma unroll
    for (int g = 0; g < 4; ++g)
#pragma unroll
      for (int i = 0; i < 8; ++i) acc[f][g][i] *= scale;
    long long mb = m0 + wrow * 32 + f * 16 + ((lane < 16) ? 0 : 8);
    // cross-lane row-max (a row lives at fixed VGPR idx across a 16-lane half)
#pragma unroll
    for (int i = 0; i < 8; ++i) {
      float mx = fmaxf(fmaxf(acc[f][0][i], acc[f][1][i]),
                       fmaxf(acc[f][2][i], acc[f][3][i]));
#pragma unroll
      for (int off = 1; off < 16; off <<= 1)
        mx = fmaxf(mx, __shfl_xor(mx, off, 32));
      if ((lane & 15) == 0) atomicMax(&rowmax[mb + i], f2ord(mx));
    }
#pragma unroll
    for (int g = 0; g < 4; ++g) {
      long long n = n0 + wcol * 64 + g * 16 + (lane & 15);
#pragma unroll
      for (int i = 0; i < 8; ++i)
        Sc[(mb + i) * (long long)SEQ + n] = f32_to_bf16(acc[f][g][i]);
    }
  }
}

__global__ void k_expsum(u16* __restrict__ Sc, const unsigned* __restrict__ rowmax,
                         float* __restrict__ rowsum) {
  __shared__ float red[256];
  int s = blockIdx.x, t = threadIdx.x;
  float mx = ord2f(rowmax[s]);
  uint4* p = (uint4*)(Sc + (size_t)s * SEQ + t * 16);
  uint4 a = p[0], b = p[1];
  unsigned w[8] = {a.x, a.y, a.z, a.w, b.x, b.y, b.z, b.w};
  float sum = 0.f;
#pragma unroll
  for (int i = 0; i < 8; ++i) {
    float lo = __expf(bf16_to_f32((u16)(w[i] & 0xffffu)) - mx);
    float hi = __expf(bf16_to_f32((u16)(w[i] >> 16)) - mx);
    sum += lo + hi;
    w[i] = pack2(lo, hi);
  }
  a.x = w[0]; a.y = w[1]; a.z = w[2]; a.w = w[3];
  b.x = w[4]; b.y = w[5]; b.z = w[6]; b.w = w[7];
  p[0] = a; p[1] = b;                                 // in-place: unnormalized probs
  red[t] = sum; __syncthreads();
  for (int st = 128; st > 0; st >>= 1) {
    if (t < st) red[t] += red[t + st];
    __syncthreads();
  }
  if (t == 0) rowsum[s] = red[0];
}

__global__ void k_pv(const u16* __restrict__ Pb, const u16* __restrict__ Vt,
                     const float* __restrict__ rowsum, float* __restrict__ out) {
  __shared__ u16 As[2 * BM * LDT];
  __shared__ u16 Bs[2 * BN * LDT];
  long long m0 = (long long)blockIdx.x * BM;
  long long n0 = (long long)blockIdx.y * BN;
  v8f acc[2][4];
#pragma unroll
  for (int f = 0; f < 2; ++f)
#pragma unroll
    for (int g = 0; g < 4; ++g) acc[f][g] = (v8f)0.0f;

  gemm_tiles<false>(Pb, Vt, SEQ, SEQ, SEQ, m0, n0, As, Bs, acc);

  int lane = threadIdx.x & 31, wid = threadIdx.x >> 5;
  int wrow = wid & 3, wcol = wid >> 2;
#pragma unroll
  for (int f = 0; f < 2; ++f) {
    long long mb = m0 + wrow * 32 + f * 16 + ((lane < 16) ? 0 : 8);
#pragma unroll
    for (int i = 0; i < 8; ++i) {
      float inv = 1.0f / rowsum[mb + i];
#pragma unroll
      for (int g = 0; g < 4; ++g) {
        long long n = n0 + wcol * 64 + g * 16 + (lane & 15);
        out[(mb + i) * DIM + n] = acc[f][g][i] * inv;
      }
    }
  }
}

// ---------- launch ----------
extern "C" void kernel_launch(void* const* d_in, const int* in_sizes, int n_in,
                              void* d_out, int out_size, void* d_ws, size_t ws_size,
                              hipStream_t stream) {
  const float* x  = (const float*)d_in[0];
  const float* Wq = (const float*)d_in[1];
  const float* Wk = (const float*)d_in[2];
  const float* Wv = (const float*)d_in[3];
  char* ws = (char*)d_ws;
  u16* Wt      = (u16*)(ws);                                   // 3 * 2 MB (bf16, transposed)
  u16* Qb      = (u16*)(ws + (6ull  << 20));                   // 8 MB
  u16* Kb      = (u16*)(ws + (14ull << 20));                   // 8 MB
  u16* Vt      = (u16*)(ws + (22ull << 20));                   // 8 MB (transposed)
  u16* Sc      = (u16*)(ws + (30ull << 20));                   // 32 MB scores/probs (fits in L2)
  unsigned* rm = (unsigned*)(ws + (62ull << 20));              // 16 KB row max (ordered uint)
  float* rs    = (float*)(ws + (62ull << 20) + SEQ * 4ull);    // 16 KB row sum
  float* out   = (float*)d_out;

  k_convert_w<<<dim3(DIM * DIM / 256, 3), 256, 0, stream>>>(Wq, Wk, Wv, Wt);
  k_init<<<SEQ / 256, 256, 0, stream>>>(rm);
  k_qkv<<<dim3(SEQ / BM, DIM / BN, 3), 256, 0, stream>>>(x, Wt, Qb, Kb, Vt);
  k_scores<<<dim3(SEQ / BM, SEQ / BN), 256, 0, stream>>>(Qb, Kb, Sc, rm);
  k_expsum<<<SEQ, 256, 0, stream>>>(Sc, rm, rs);
  k_pv<<<dim3(SEQ / BM, DIM / BN), 256, 0, stream>>>(Sc, Vt, rs, out);
}
